// GCN_LPA_74895639708067
// MI455X (gfx1250) — matrix operand order repaired and checked by reading, not compiled, not run on hardware
//
#include <hip/hip_runtime.h>
#include <hip/hip_bf16.h>

#define N_NODES 50000
#define N_EDGES 800000
#define IN_C    512
#define HID_C   256
#define OUT_C   40
#define OUT_PAD 128
#define BN_EPSF 1e-5f

typedef __attribute__((ext_vector_type(16))) __bf16 v16bf;
typedef __attribute__((ext_vector_type(8)))  float  v8f;

// ---------------------------------------------------------------- utilities
__global__ void fill_f32_k(float* __restrict__ p, float v, long long n) {
    long long i = (long long)blockIdx.x * blockDim.x + threadIdx.x;
    if (i < n) p[i] = v;
}

__global__ void convert_bf16_k(const float* __restrict__ in, __bf16* __restrict__ out, long long n) {
    long long i = (long long)blockIdx.x * blockDim.x + threadIdx.x;
    if (i < n) out[i] = (__bf16)in[i];
}

// Wt[n][k] = bf16(W[k][n]); rows n >= N zero padded up to Npad
__global__ void transpose_w_k(const float* __restrict__ W, __bf16* __restrict__ Wt,
                              int K, int N, int Npad) {
    long long idx = (long long)blockIdx.x * blockDim.x + threadIdx.x;
    long long total = (long long)Npad * K;
    if (idx >= total) return;
    int n = (int)(idx / K);
    int k = (int)(idx - (long long)n * K);
    float v = (n < N) ? W[(size_t)k * N + n] : 0.f;
    Wt[idx] = (__bf16)v;
}

// ---------------------------------------------------------------- gcn norm
__global__ void deg_scatter_k(const int* __restrict__ row, const float* __restrict__ w,
                              float* __restrict__ deg, int E) {
    int e = blockIdx.x * blockDim.x + threadIdx.x;
    if (e < E) unsafeAtomicAdd(&deg[row[e]], w[e]);
}

__global__ void finish_norm_k(float* __restrict__ dinv_io, float* __restrict__ selfnorm, int n) {
    int i = blockIdx.x * blockDim.x + threadIdx.x;
    if (i >= n) return;
    float d = dinv_io[i];           // deg (>= 1 due to self loop weight 1)
    float r = rsqrtf(d);
    dinv_io[i] = r;
    selfnorm[i] = r * r;            // dinv * 1 * dinv
}

__global__ void edge_norm_k(const int* __restrict__ row, const int* __restrict__ col,
                            const float* __restrict__ w, const float* __restrict__ dinv,
                            float* __restrict__ nrm, int E) {
    int e = blockIdx.x * blockDim.x + threadIdx.x;
    if (e < E) nrm[e] = dinv[row[e]] * w[e] * dinv[col[e]];
}

// ---------------------------------------------------------------- WMMA GEMM
// C[M,Nout] = A[M,K](bf16) * Wt[Npad,K](bf16, pre-transposed), f32 accumulate.
// Optionally also writes H[r,c] = selfnorm[r]*C[r,c] + bias[c] (fused self-loop init).
// block = 256 threads = 8 waves; block tile 128x128; wave: 16 rows x 128 cols.
// LDS row stride 40 bf16 = 80B -> all fragment chunks 16B aligned (ds_load_b128),
// and 20-bank stride gives conflict-free row access. Double-buffered K tiles.
#define LSTR 40

__global__ __launch_bounds__(256)
void gemm_bf16_wmma_k(const __bf16* __restrict__ A, const __bf16* __restrict__ Bt,
                      float* __restrict__ C, float* __restrict__ H,
                      const float* __restrict__ selfnorm, const float* __restrict__ bias,
                      int M, int K, int Nout) {
    __shared__ __align__(16) __bf16 Al[2][128 * LSTR];
    __shared__ __align__(16) __bf16 Bl[2][128 * LSTR];

    const int tid  = threadIdx.x;
    const int lane = tid & 31;
    const int wid  = tid >> 5;
    const int mb   = blockIdx.x * 128;
    const int nb   = blockIdx.y * 128;

    // tile staging: 512 uint4 per (A|B) tile, 2 per thread: row = p>>2, quarter = p&3
    const int r0 = tid >> 2;          // 0..63
    const int r1 = r0 + 64;           // 64..127
    const int q8 = (tid & 3) * 8;     // bf16 offset of 16B quarter within 32-k row

    v8f acc[8];
#pragma unroll
    for (int i = 0; i < 8; ++i)
#pragma unroll
        for (int j = 0; j < 8; ++j) acc[i][j] = 0.f;

    uint4 as0, as1, bs0, bs1;
    const uint4 z4 = make_uint4(0u, 0u, 0u, 0u);

    auto load_stage = [&](int kc) {
        int gr0 = mb + r0, gr1 = mb + r1;
        as0 = (gr0 < M) ? *(const uint4*)(A + (size_t)gr0 * K + kc + q8) : z4;
        as1 = (gr1 < M) ? *(const uint4*)(A + (size_t)gr1 * K + kc + q8) : z4;
        bs0 = *(const uint4*)(Bt + (size_t)(nb + r0) * K + kc + q8);
        bs1 = *(const uint4*)(Bt + (size_t)(nb + r1) * K + kc + q8);
    };
    auto store_stage = [&](int buf) {
        *(uint4*)(&Al[buf][r0 * LSTR + q8]) = as0;
        *(uint4*)(&Al[buf][r1 * LSTR + q8]) = as1;
        *(uint4*)(&Bl[buf][r0 * LSTR + q8]) = bs0;
        *(uint4*)(&Bl[buf][r1 * LSTR + q8]) = bs1;
    };

    load_stage(0);
    store_stage(0);
    __syncthreads();

    const int nsteps = K >> 5;
    int cur = 0;
    for (int s = 0; s < nsteps; ++s) {
        if (s + 1 < nsteps) load_stage((s + 1) << 5);

        // A fragment (16x32 bf16, wave32): lane m = lane&15, kb = (lane>>4)*8;
        // VGPR0-3 = k[kb..kb+7], VGPR4-7 = k[16+kb..16+kb+7] -> two 16B chunks
        union { v16bf v; uint4 q[2]; } af;
        const __bf16* ar = &Al[cur][(wid * 16 + (lane & 15)) * LSTR + ((lane >> 4) << 3)];
        af.q[0] = *(const uint4*)(ar);
        af.q[1] = *(const uint4*)(ar + 16);

#pragma unroll
        for (int nt = 0; nt < 8; ++nt) {
            // B fragment (32x16): lane n = lane&15, kb = (lane>>4)*8; k[kb..kb+7], k[16+kb..16+kb+7]
            union { v16bf v; uint4 q[2]; } bf_;
            const __bf16* br = &Bl[cur][(nt * 16 + (lane & 15)) * LSTR + ((lane >> 4) << 3)];
            bf_.q[0] = *(const uint4*)(br);
            bf_.q[1] = *(const uint4*)(br + 16);
            acc[nt] = __builtin_amdgcn_wmma_f32_16x16x32_bf16(
                false, af.v, false, bf_.v, (short)0, acc[nt], false, false);
        }

        if (s + 1 < nsteps) {
            int nxt = cur ^ 1;
            store_stage(nxt);
            __syncthreads();
            cur = nxt;
        }
    }

    // store: C/D layout -> M = (lane>>4)*8 + r within 16-row tile, N = lane&15
#pragma unroll
    for (int nt = 0; nt < 8; ++nt) {
        int coln = nb + nt * 16 + (lane & 15);
        if (coln >= Nout) continue;
#pragma unroll
        for (int r = 0; r < 8; ++r) {
            int row = mb + wid * 16 + ((lane >> 4) << 3) + r;
            if (row < M) {
                float v = acc[nt][r];
                C[(size_t)row * Nout + coln] = v;
                if (H) H[(size_t)row * Nout + coln] =
                        selfnorm[row] * v + (bias ? bias[coln] : 0.f);
            }
        }
    }
}

// ---------------------------------------------------------------- propagate
// out[r,c] = selfnorm[r]*z[r,c]   (self loop init, LPA path)
__global__ void prop_init_k(const float* __restrict__ z, const float* __restrict__ selfnorm,
                            float* __restrict__ out, int nrows, int C) {
    long long idx = (long long)blockIdx.x * blockDim.x + threadIdx.x;
    long long total = (long long)nrows * C;
    if (idx >= total) return;
    int r = (int)(idx / C);
    out[idx] = selfnorm[r] * z[idx];
}

// scatter: out[row[e], :] += norm[e] * z[col[e], :]   (float4 per thread)
__global__ void spmm_scatter_k(const float* __restrict__ z, const int* __restrict__ rows,
                               const int* __restrict__ cols, const float* __restrict__ nrm,
                               float* __restrict__ out, int E, int C) {
    int chunks = C >> 2;
    long long idx = (long long)blockIdx.x * blockDim.x + threadIdx.x;
    long long total = (long long)E * chunks;
    if (idx >= total) return;
    int e = (int)(idx / chunks);
    int q = (int)(idx - (long long)e * chunks);
    int r = rows[e], c = cols[e];
    float w = nrm[e];
    const float4 v = *(const float4*)(z + (size_t)c * C + (size_t)q * 4);
    float* o = out + (size_t)r * C + (size_t)q * 4;
    unsafeAtomicAdd(o + 0, w * v.x);
    unsafeAtomicAdd(o + 1, w * v.y);
    unsafeAtomicAdd(o + 2, w * v.z);
    unsafeAtomicAdd(o + 3, w * v.w);
}

// ---------------------------------------------------------------- batchnorm
__global__ __launch_bounds__(256)
void bn_stats_k(const float* __restrict__ h, float* __restrict__ stats, int nrows) {
    int c  = threadIdx.x;                 // HID_C threads
    int r0 = blockIdx.x * 512;
    int r1 = min(r0 + 512, nrows);
    float s = 0.f, s2 = 0.f;
    for (int r = r0; r < r1; ++r) {
        float v = h[(size_t)r * HID_C + c];
        s += v; s2 += v * v;
    }
    unsafeAtomicAdd(&stats[c], s);
    unsafeAtomicAdd(&stats[HID_C + c], s2);
}

__global__ void bn_finalize_k(float* __restrict__ stats, const float* __restrict__ g,
                              const float* __restrict__ beta, int nrows) {
    int c = threadIdx.x;
    float inv_n = 1.f / (float)nrows;
    float mu  = stats[c] * inv_n;
    float var = stats[HID_C + c] * inv_n - mu * mu;
    var = fmaxf(var, 0.f);
    float scale = g[c] * rsqrtf(var + BN_EPSF);
    stats[c] = scale;
    stats[HID_C + c] = beta[c] - mu * scale;   // shift
}

__global__ void bn_apply_relu_bf16_k(const float* __restrict__ h, const float* __restrict__ stats,
                                     __bf16* __restrict__ out, long long total) {
    long long i = (long long)blockIdx.x * blockDim.x + threadIdx.x;
    if (i >= total) return;
    int c = (int)(i & (HID_C - 1));
    float v = h[i] * stats[c] + stats[HID_C + c];
    out[i] = (__bf16)fmaxf(v, 0.f);
}

// ---------------------------------------------------------------- launcher
static inline int cdiv_i(long long a, long long b) { return (int)((a + b - 1) / b); }

extern "C" void kernel_launch(void* const* d_in, const int* in_sizes, int n_in,
                              void* d_out, int out_size, void* d_ws, size_t ws_size,
                              hipStream_t stream) {
    const float* x      = (const float*)d_in[0];
    const int*   erow   = (const int*)d_in[1];
    const int*   ecol   = erow + N_EDGES;
    const float* ylab   = (const float*)d_in[2];
    const float* ylab2  = (const float*)d_in[3];
    const float* ew     = (const float*)d_in[4];
    const float* W0     = (const float*)d_in[5];
    const float* b0     = (const float*)d_in[6];
    const float* W1     = (const float*)d_in[7];
    const float* b1     = (const float*)d_in[8];
    const float* W2     = (const float*)d_in[9];
    const float* b2     = (const float*)d_in[10];
    const float* g0     = (const float*)d_in[11];
    const float* beta0  = (const float*)d_in[12];
    const float* g1     = (const float*)d_in[13];
    const float* beta1  = (const float*)d_in[14];
    float* out = (float*)d_out;

    // workspace carve-up
    char* ws = (char*)d_ws;
    size_t off = 0;
    auto take = [&](size_t bytes) -> void* {
        void* p = ws + off;
        off = (off + bytes + 255) & ~(size_t)255;
        return p;
    };
    float*  dinv     = (float*)take((size_t)N_NODES * 4);
    float*  selfnorm = (float*)take((size_t)N_NODES * 4);
    float*  normw    = (float*)take((size_t)N_EDGES * 4);
    __bf16* W0t      = (__bf16*)take((size_t)HID_C * IN_C * 2);
    __bf16* W1t      = (__bf16*)take((size_t)HID_C * HID_C * 2);
    __bf16* W2t      = (__bf16*)take((size_t)OUT_PAD * HID_C * 2);
    float*  stats    = (float*)take((size_t)2 * HID_C * 4);
    __bf16* abf      = (__bf16*)take((size_t)N_NODES * IN_C * 2);   // x_bf / act_bf / LPA pong
    float*  bufZ     = (float*)take((size_t)N_NODES * HID_C * 4);   // GEMM out (raw z)
    float*  bufH     = (float*)take((size_t)N_NODES * HID_C * 4);   // propagate out / LPA ping

    const int T = 256;

    // ---- conversions
    convert_bf16_k<<<cdiv_i((long long)N_NODES * IN_C, T), T, 0, stream>>>(x, abf, (long long)N_NODES * IN_C);
    transpose_w_k<<<cdiv_i((long long)HID_C * IN_C, T), T, 0, stream>>>(W0, W0t, IN_C, HID_C, HID_C);
    transpose_w_k<<<cdiv_i((long long)HID_C * HID_C, T), T, 0, stream>>>(W1, W1t, HID_C, HID_C, HID_C);
    transpose_w_k<<<cdiv_i((long long)OUT_PAD * HID_C, T), T, 0, stream>>>(W2, W2t, HID_C, OUT_C, OUT_PAD);

    // ---- gcn normalization
    fill_f32_k<<<cdiv_i(N_NODES, T), T, 0, stream>>>(dinv, 1.0f, N_NODES);  // self-loop weight
    deg_scatter_k<<<cdiv_i(N_EDGES, T), T, 0, stream>>>(erow, ew, dinv, N_EDGES);
    finish_norm_k<<<cdiv_i(N_NODES, T), T, 0, stream>>>(dinv, selfnorm, N_NODES);
    edge_norm_k<<<cdiv_i(N_EDGES, T), T, 0, stream>>>(erow, ecol, ew, dinv, normw, N_EDGES);

    // ---- layer 0: z = x@W0 (+fused h-init) ; h += edge scatter ; BN+ReLU -> abf
    {
        dim3 g(cdiv_i(N_NODES, 128), HID_C / 128);
        gemm_bf16_wmma_k<<<g, T, 0, stream>>>(abf, W0t, bufZ, bufH, selfnorm, b0, N_NODES, IN_C, HID_C);
    }
    spmm_scatter_k<<<cdiv_i((long long)N_EDGES * (HID_C / 4), T), T, 0, stream>>>(bufZ, erow, ecol, normw, bufH, N_EDGES, HID_C);
    fill_f32_k<<<cdiv_i(2 * HID_C, T), T, 0, stream>>>(stats, 0.f, 2 * HID_C);
    bn_stats_k<<<cdiv_i(N_NODES, 512), HID_C, 0, stream>>>(bufH, stats, N_NODES);
    bn_finalize_k<<<1, HID_C, 0, stream>>>(stats, g0, beta0, N_NODES);
    bn_apply_relu_bf16_k<<<cdiv_i((long long)N_NODES * HID_C, T), T, 0, stream>>>(bufH, stats, abf, (long long)N_NODES * HID_C);

    // ---- layer 1
    {
        dim3 g(cdiv_i(N_NODES, 128), HID_C / 128);
        gemm_bf16_wmma_k<<<g, T, 0, stream>>>(abf, W1t, bufZ, bufH, selfnorm, b1, N_NODES, HID_C, HID_C);
    }
    spmm_scatter_k<<<cdiv_i((long long)N_EDGES * (HID_C / 4), T), T, 0, stream>>>(bufZ, erow, ecol, normw, bufH, N_EDGES, HID_C);
    fill_f32_k<<<cdiv_i(2 * HID_C, T), T, 0, stream>>>(stats, 0.f, 2 * HID_C);
    bn_stats_k<<<cdiv_i(N_NODES, 512), HID_C, 0, stream>>>(bufH, stats, N_NODES);
    bn_finalize_k<<<1, HID_C, 0, stream>>>(stats, g1, beta1, N_NODES);
    bn_apply_relu_bf16_k<<<cdiv_i((long long)N_NODES * HID_C, T), T, 0, stream>>>(bufH, stats, abf, (long long)N_NODES * HID_C);

    // ---- layer 2: z40 = h@W2 (Npad=128, store 40 cols, fused out-init) ; out += scatter
    {
        dim3 g(cdiv_i(N_NODES, 128), OUT_PAD / 128);
        gemm_bf16_wmma_k<<<g, T, 0, stream>>>(abf, W2t, bufZ, out, selfnorm, b2, N_NODES, HID_C, OUT_C);
    }
    spmm_scatter_k<<<cdiv_i((long long)N_EDGES * (OUT_C / 4), T), T, 0, stream>>>(bufZ, erow, ecol, normw, out, N_EDGES, OUT_C);

    // ---- label propagation (3 iters each, two label matrices)
    const float* labs[2] = { ylab, ylab2 };
    for (int m = 0; m < 2; ++m) {
        float* dst = out + (size_t)(m + 1) * N_NODES * OUT_C;
        float* ping = bufH;
        float* pong = (float*)abf;
        const float* cur = labs[m];
        for (int it = 0; it < 3; ++it) {
            float* o = (it == 2) ? dst : ((it == 0) ? ping : pong);
            prop_init_k<<<cdiv_i((long long)N_NODES * OUT_C, T), T, 0, stream>>>(cur, selfnorm, o, N_NODES, OUT_C);
            spmm_scatter_k<<<cdiv_i((long long)N_EDGES * (OUT_C / 4), T), T, 0, stream>>>(cur, erow, ecol, normw, o, N_EDGES, OUT_C);
            cur = o;
        }
    }
}
